// Mymodel_33947421508359
// MI455X (gfx1250) — compile-verified, matrix operand order checked by reading, not run on hardware
//
#include <hip/hip_runtime.h>
#include <math.h>

// ---------------------------------------------------------------------------
// BiLSTM-CRF loss for MI455X (gfx1250, wave32, WMMA 16x16x32 bf16 -> f32,
// Tensor Data Mover for LDS staging).
// B=128, L=512, E=H=256, 4H=1024, T=16, BL=65536.
// ---------------------------------------------------------------------------

typedef __bf16 bf16;
typedef __attribute__((ext_vector_type(16))) __bf16 bf16x16;
typedef __attribute__((ext_vector_type(8)))  __bf16 bf16x8;
typedef __attribute__((ext_vector_type(8)))  float  f32x8;
typedef __attribute__((ext_vector_type(4)))  unsigned int u32x4;
typedef __attribute__((ext_vector_type(4)))  int    i32x4;
typedef __attribute__((ext_vector_type(8)))  int    i32x8;

#define BATCH 128
#define SEQ   512
#define EMB   256
#define HID   256
#define GATES 1024   // 4*HID
#define NTAG  16
#define BL    65536  // BATCH*SEQ
#define TSTART 14
#define TEND   15

__device__ __forceinline__ f32x8 wmma_bf16(bf16x16 a, bf16x16 b, f32x8 c) {
    return __builtin_amdgcn_wmma_f32_16x16x32_bf16(false, a, false, b,
                                                   (short)0, c, false, false);
}

// ---------------------------------------------------------------------------
// Tensor Data Mover: DMA a 2D tile (rows x dim0 elements of 2 bytes, row
// stride `stride` elements) from global memory into LDS.  D# per CDNA5 ISA
// ch.8: group0 = {count=1, lds_addr, global_addr[56:0], type=2};
// group1 = {mask=0, data_size=2B, tensor_dim0/1, tile_dim0/1, dim0_stride}.
// Issued once per wave (descriptor is scalar); completion via TENSORcnt.
// ---------------------------------------------------------------------------
__device__ __forceinline__ void tdm_load_to_lds(void* lds_ptr, const void* gptr,
                                                unsigned dim0, unsigned rows,
                                                unsigned stride) {
    unsigned long long ga = (unsigned long long)(uintptr_t)gptr;
    unsigned la = (unsigned)(uintptr_t)lds_ptr;     // flat addr[31:0] = LDS byte addr
    u32x4 g0 = { 1u,                                 // count=1, user descriptor
                 la,
                 (unsigned)ga,
                 (unsigned)((ga >> 32) & 0x01FFFFFFu) | (2u << 30) };  // type=2
    i32x8 g1 = { (int)0x00010000u,                   // wg_mask=0, data_size=2B
                 (int)((dim0 & 0xFFFFu) << 16),      // tensor_dim0 lo16
                 (int)((dim0 >> 16) | ((rows & 0xFFFFu) << 16)),   // dim0 hi | dim1 lo
                 (int)((rows >> 16) | (dim0 << 16)), // dim1 hi | tile_dim0
                 (int)(rows & 0xFFFFu),              // tile_dim1 (tile_dim2=0)
                 (int)stride,                        // tensor_dim0_stride lo32
                 0, 0 };
    i32x4 z4 = {0, 0, 0, 0};
#if __clang_major__ >= 23
    i32x8 z8 = {0, 0, 0, 0, 0, 0, 0, 0};
    __builtin_amdgcn_tensor_load_to_lds(g0, g1, z4, z4, z8, 0);
#else
    __builtin_amdgcn_tensor_load_to_lds(g0, g1, z4, z4, 0);
#endif
}

// ---------------------------------------------------------------------------
// 1) Embedding gather + convert to bf16.  x_bf[m][e], m = b*SEQ + t.
// ---------------------------------------------------------------------------
__global__ void gather_embed(const int* __restrict__ ids,
                             const float* __restrict__ emb,
                             bf16* __restrict__ xbf) {
    size_t q = (size_t)blockIdx.x * blockDim.x + threadIdx.x;
    size_t e4 = q * 4;
    if (e4 >= (size_t)BL * EMB) return;
    size_t m = e4 >> 8;
    int    c = (int)(e4 & 255);
    const float* src = emb + (size_t)ids[m] * EMB + c;
    bf16*        dst = xbf + m * EMB + c;
#pragma unroll
    for (int i = 0; i < 4; ++i) dst[i] = (bf16)src[i];
}

__global__ void cvt_f32_bf16(const float* __restrict__ src,
                             bf16* __restrict__ dst, int n) {
    int i = blockIdx.x * blockDim.x + threadIdx.x;
    if (i < n) dst[i] = (bf16)src[i];
}

__global__ void zero_out(float* o) { *o = 0.0f; }

// ---------------------------------------------------------------------------
// 2) Input-projection GEMM:  M=65536, N=1024/dir, K=256.
//    Block = 256 thr (8 waves), 16x128 tile; A tile staged via TDM.
//    Output layout (recurrence-friendly): Xproj[dir][t][chunk][n][16 batch].
// ---------------------------------------------------------------------------
__global__ void xproj_gemm(const bf16* __restrict__ xbf,
                           const bf16* __restrict__ wih,   // [2][1024][256]
                           const float* __restrict__ bias_f,
                           const float* __restrict__ bias_b,
                           bf16* __restrict__ xproj) {
    const int mt   = blockIdx.x;          // 0..4095
    const int nb   = blockIdx.y;          // 0..7
    const int dir  = blockIdx.z;          // 0..1
    const int w    = threadIdx.x >> 5;
    const int lane = threadIdx.x & 31;
    const int khalf = lane >> 4, nl = lane & 15;

    __shared__ __attribute__((aligned(32))) bf16 atile[16][256];
    const int m0 = mt * 16;
    if (w == 0) {   // wave 0 issues the tile DMA (EXEC-independent, per-wave)
        tdm_load_to_lds(&atile[0][0], xbf + (size_t)m0 * EMB, 256u, 16u, 256u);
        __builtin_amdgcn_s_wait_tensorcnt(0);
    }
    __syncthreads();

    const int n0 = nb * 128 + w * 16;
    const bf16*  wd   = wih + (size_t)dir * GATES * EMB;
    const float* bias = dir ? bias_b : bias_f;

    f32x8 acc = {0.f, 0.f, 0.f, 0.f, 0.f, 0.f, 0.f, 0.f};
    const int n = n0 + nl;
#pragma unroll
    for (int k = 0; k < 8; ++k) {
        bf16x16 a  = *(const bf16x16*)&atile[nl][32 * k + 16 * khalf];
        bf16x16 bm = *(const bf16x16*)&wd[(size_t)n * EMB + 32 * k + 16 * khalf];
        acc = wmma_bf16(a, bm, acc);
    }
    const float bv = bias[n];
#pragma unroll
    for (int r = 0; r < 8; ++r) {
        int m  = m0 + r + 8 * khalf;          // C layout: M = r (+8 for hi lanes)
        int bb = m >> 9, t = m & 511;
        int chunk = bb >> 4, bi = bb & 15;
        xproj[((((size_t)dir * SEQ + t) * 8 + chunk) * (GATES * 16))
              + (size_t)n * 16 + bi] = (bf16)(acc[r] + bv);
    }
}

// ---------------------------------------------------------------------------
// 3) Recurrent LSTM.  16 blocks: blockIdx.x = dir*8 + batch_chunk(16 rows).
//    512 threads = 16 waves; wave w owns h-cols [16w,16w+16) => gate tiles
//    i,f,g,o at n = g*256 + 16w.  W_hh fragments are REGISTER-RESIDENT for
//    all 512 steps (32 x bf16x16 = 256 VGPRs/wave); c stays in a v8f
//    accumulator; h ping-pongs in LDS; the per-step 32 KB Xproj slice is
//    double-buffered in LDS via TDM, prefetched one step ahead.
// ---------------------------------------------------------------------------
__global__ void lstm_kernel(const bf16* __restrict__ xproj,  // [2][512][8][1024][16]
                            const bf16* __restrict__ whh,    // [2][1024][256]
                            bf16* __restrict__ hout) {       // [BL][512] (hf|hb)
    const int dir   = blockIdx.x >> 3;
    const int chunk = blockIdx.x & 7;
    const int w     = threadIdx.x >> 5;
    const int lane  = threadIdx.x & 31;
    const int khalf = lane >> 4, nl = lane & 15;
    const int b0    = chunk * 16;

    __shared__ __attribute__((aligned(32))) bf16 hbuf[2][16][HID];     // 16 KB
    __shared__ __attribute__((aligned(32))) bf16 xstage[2][GATES * 16]; // 64 KB
    for (int i = threadIdx.x; i < 2 * 16 * HID; i += 512)
        ((bf16*)hbuf)[i] = (bf16)0.0f;

    const bf16* wd = whh + (size_t)dir * GATES * HID;
    const int hcol = w * 16 + nl;

    // W_hh fragments: resident in VGPRs for the whole sequence.
    bf16x16 wfrag[4][8];
#pragma unroll
    for (int g = 0; g < 4; ++g) {
        const int n = g * 256 + hcol;
#pragma unroll
        for (int k = 0; k < 8; ++k)
            wfrag[g][k] = *(const bf16x16*)&wd[(size_t)n * HID + 32 * k + 16 * khalf];
    }

    // Stage t=0's gate pre-activations.
    {
        int t0 = dir ? (SEQ - 1) : 0;
        if (w == 0) {
            tdm_load_to_lds(&xstage[0][0],
                            xproj + (((size_t)dir * SEQ + t0) * 8 + chunk) * (GATES * 16),
                            (unsigned)(GATES * 16), 1u, (unsigned)(GATES * 16));
            __builtin_amdgcn_s_wait_tensorcnt(0);
        }
    }
    __syncthreads();

    f32x8 c = {0.f, 0.f, 0.f, 0.f, 0.f, 0.f, 0.f, 0.f};

    for (int it = 0; it < SEQ; ++it) {
        const int t   = dir ? (SEQ - 1 - it) : it;
        const int cur = it & 1, nxt = cur ^ 1;

        // Kick off DMA for the next timestep's slice into the other buffer.
        if (it < SEQ - 1 && w == 0) {
            int tn = dir ? (SEQ - 2 - it) : (it + 1);
            tdm_load_to_lds(&xstage[nxt][0],
                            xproj + (((size_t)dir * SEQ + tn) * 8 + chunk) * (GATES * 16),
                            (unsigned)(GATES * 16), 1u, (unsigned)(GATES * 16));
        }

        // acc init from staged Xproj: one contiguous 16B LDS read per gate.
        f32x8 acc[4];
#pragma unroll
        for (int g = 0; g < 4; ++g) {
            const int n = g * 256 + hcol;
            bf16x8 xv = *(const bf16x8*)&xstage[cur][n * 16 + 8 * khalf];
#pragma unroll
            for (int r = 0; r < 8; ++r) acc[g][r] = (float)xv[r];
        }
        // Recurrent GEMM: h_{t-1} (LDS) x W_hh (registers).
#pragma unroll
        for (int k = 0; k < 8; ++k) {
            bf16x16 a = *(const bf16x16*)&hbuf[cur][nl][32 * k + 16 * khalf];
#pragma unroll
            for (int g = 0; g < 4; ++g)
                acc[g] = wmma_bf16(a, wfrag[g][k], acc[g]);
        }
        __syncthreads();   // all reads of hbuf[cur]/xstage[cur] done

#pragma unroll
        for (int r = 0; r < 8; ++r) {
            float ig = 1.f / (1.f + expf(-acc[0][r]));
            float fg = 1.f / (1.f + expf(-acc[1][r]));
            float gg = tanhf(acc[2][r]);
            float og = 1.f / (1.f + expf(-acc[3][r]));
            c[r] = fg * c[r] + ig * gg;
            float h = og * tanhf(c[r]);
            int mrow = r + 8 * khalf;
            hbuf[nxt][mrow][hcol] = (bf16)h;
            int bg = b0 + mrow;
            hout[((size_t)bg * SEQ + t) * (2 * HID) + dir * HID + hcol] = (bf16)h;
        }
        // Next buffer must be fully DMA'd before anyone reads it.
        if (it < SEQ - 1 && w == 0) __builtin_amdgcn_s_wait_tensorcnt(0);
        __syncthreads();
    }
}

// ---------------------------------------------------------------------------
// 4) Classifier: emission[m][:16] = hcat[m] @ Wcls^T + bcls.  K=512 -> 16 WMMAs.
// ---------------------------------------------------------------------------
__global__ void cls_gemm(const bf16* __restrict__ hout,   // [BL][512]
                         const bf16* __restrict__ wcls,   // [16][512]
                         const float* __restrict__ bcls,  // [16]
                         float* __restrict__ emis) {      // [BL][16]
    const int m0   = blockIdx.x * 16;
    const int lane = threadIdx.x & 31;
    const int khalf = lane >> 4, nl = lane & 15;

    const float bv = bcls[nl];
    f32x8 acc = {bv, bv, bv, bv, bv, bv, bv, bv};
#pragma unroll
    for (int k = 0; k < 16; ++k) {
        bf16x16 a  = *(const bf16x16*)&hout[(size_t)(m0 + nl) * 512 + 32 * k + 16 * khalf];
        bf16x16 bm = *(const bf16x16*)&wcls[(size_t)nl * 512 + 32 * k + 16 * khalf];
        acc = wmma_bf16(a, bm, acc);
    }
#pragma unroll
    for (int r = 0; r < 8; ++r) {
        int m = m0 + r + 8 * khalf;
        emis[(size_t)m * NTAG + nl] = acc[r];
    }
}

// ---------------------------------------------------------------------------
// 5) CRF: golden path score + forward log-sum-exp recursion, one wave per b.
// ---------------------------------------------------------------------------
__global__ void crf_kernel(const int* __restrict__ tags,
                           const float* __restrict__ emis,   // [BL][16]
                           const float* __restrict__ trans,  // [16][16]
                           float* __restrict__ out) {
    const int b    = blockIdx.x;
    const int lane = threadIdx.x;
    const int j    = lane & 15;

    __shared__ int   len_s;
    __shared__ float gold_s;
    if (lane == 0) len_s = 0;
    __syncthreads();
    int part = 0;
    for (int t = lane; t < SEQ; t += 32) part += (tags[b * SEQ + t] != 0);
    atomicAdd(&len_s, part);
    __syncthreads();
    const int len = len_s;

    if (lane == 0) {
        float g = 0.f; int prev = TSTART;
        for (int t = 0; t < SEQ; ++t) {
            int tg = tags[b * SEQ + t];
            if (tg != 0) {
                g += emis[((size_t)b * SEQ + t) * NTAG + tg] + trans[prev * NTAG + tg];
                prev = tg;
            }
        }
        gold_s = g;
    }

    float tr[NTAG];
#pragma unroll
    for (int i = 0; i < NTAG; ++i) tr[i] = trans[i * NTAG + j];

    float alpha = emis[(size_t)b * SEQ * NTAG + j] + trans[TSTART * NTAG + j];
    for (int t = 1; t < SEQ; ++t) {
        float vals[NTAG], mx = -3.4e38f;
#pragma unroll
        for (int i = 0; i < NTAG; ++i) {
            float ai = __shfl(alpha, i, 32);
            vals[i] = ai + tr[i];
            mx = fmaxf(mx, vals[i]);
        }
        float s = 0.f;
#pragma unroll
        for (int i = 0; i < NTAG; ++i) s += expf(vals[i] - mx);
        float nv = mx + logf(s) + emis[((size_t)b * SEQ + t) * NTAG + j];
        if (t < len) alpha = nv;
    }
    __syncthreads();
    float fin = __shfl(alpha, TEND, 32);
    if (lane == 0) atomicAdd(out, (fin - gold_s) * (1.0f / (float)BATCH));
}

// ---------------------------------------------------------------------------
extern "C" void kernel_launch(void* const* d_in, const int* in_sizes, int n_in,
                              void* d_out, int out_size, void* d_ws, size_t ws_size,
                              hipStream_t stream) {
    (void)in_sizes; (void)n_in; (void)out_size; (void)ws_size;
    const int*   ids   = (const int*)d_in[0];
    const int*   tags  = (const int*)d_in[1];
    const float* emb   = (const float*)d_in[2];
    const float* wihf  = (const float*)d_in[3];
    const float* whhf  = (const float*)d_in[4];
    const float* bf_   = (const float*)d_in[5];
    const float* wihb  = (const float*)d_in[6];
    const float* whhb  = (const float*)d_in[7];
    const float* bb_   = (const float*)d_in[8];
    const float* wcls  = (const float*)d_in[9];
    const float* bcls  = (const float*)d_in[10];
    const float* trans = (const float*)d_in[11];

    char* ws = (char*)d_ws; size_t off = 0;
    auto carve = [&](size_t bytes) -> char* {
        char* p = ws + off; off = (off + bytes + 255) & ~(size_t)255; return p;
    };
    bf16* xbf   = (bf16*)carve((size_t)BL * EMB * 2);                 //  33.5 MB
    bf16* wihB  = (bf16*)carve((size_t)2 * GATES * EMB * 2);          //   1 MB
    bf16* whhB  = (bf16*)carve((size_t)2 * GATES * HID * 2);          //   1 MB
    bf16* wclsB = (bf16*)carve((size_t)NTAG * 512 * 2);               //  16 KB
    bf16* xproj = (bf16*)carve((size_t)2 * SEQ * BATCH * GATES * 2);  // 268 MB
    bf16* hout  = (bf16*)carve((size_t)BL * 512 * 2);                 //  67 MB
    float* emis = (float*)carve((size_t)BL * NTAG * 4);               //   4 MB

    gather_embed<<<16384, 256, 0, stream>>>(ids, emb, xbf);
    cvt_f32_bf16<<<1024, 256, 0, stream>>>(wihf, wihB, GATES * EMB);
    cvt_f32_bf16<<<1024, 256, 0, stream>>>(wihb, wihB + (size_t)GATES * EMB, GATES * EMB);
    cvt_f32_bf16<<<1024, 256, 0, stream>>>(whhf, whhB, GATES * HID);
    cvt_f32_bf16<<<1024, 256, 0, stream>>>(whhb, whhB + (size_t)GATES * HID, GATES * HID);
    cvt_f32_bf16<<<32, 256, 0, stream>>>(wcls, wclsB, NTAG * 512);

    dim3 g1(BL / 16, 8, 2);
    xproj_gemm<<<g1, 256, 0, stream>>>(xbf, wihB, bf_, bb_, xproj);
    lstm_kernel<<<16, 512, 0, stream>>>(xproj, whhB, hout);
    cls_gemm<<<BL / 16, 32, 0, stream>>>(hout, wclsB, bcls, emis);
    zero_out<<<1, 1, 0, stream>>>((float*)d_out);
    crf_kernel<<<BATCH, 32, 0, stream>>>(tags, emis, trans, (float*)d_out);
}
